// VectorQuantizer_45268955299923
// MI455X (gfx1250) — compile-verified, matrix operand order checked by reading, not run on hardware
//
#include <hip/hip_runtime.h>

typedef __attribute__((ext_vector_type(16))) __bf16 bf16x16;
typedef __attribute__((ext_vector_type(8)))  __bf16 bf16x8;
typedef __attribute__((ext_vector_type(8)))  float  f32x8;

constexpr int D      = 512;
constexpr int K      = 4096;
constexpr int NROWS  = 64 * 512;          // 32768 query rows
constexpr int BM     = 256;               // rows per WG: 8 waves x 32 rows
constexpr int NQ     = NROWS * D;         // 16777216 quantized elements
constexpr int LDB    = 520;               // padded LDS row stride (bank-conflict free)
constexpr int NCHUNK = K / 16;            // 256 chunks of 16 codes

// ---------------------------------------------------------------------------
// Kernel 1: codebook norms (exact fp32) + bf16 codebook copy + zero loss acc
// ---------------------------------------------------------------------------
__global__ void vq_prep(const float* __restrict__ cb,
                        float* __restrict__ cnorm,
                        __bf16* __restrict__ cbh,
                        float* __restrict__ lossAcc) {
    const int k   = blockIdx.x;
    const int tid = threadIdx.x;            // 64 threads
    const int i   = tid * 8;
    const float4* p = (const float4*)(cb + (size_t)k * D + i);
    float4 f0 = p[0], f1 = p[1];

    bf16x8 h;
    h[0] = (__bf16)f0.x; h[1] = (__bf16)f0.y; h[2] = (__bf16)f0.z; h[3] = (__bf16)f0.w;
    h[4] = (__bf16)f1.x; h[5] = (__bf16)f1.y; h[6] = (__bf16)f1.z; h[7] = (__bf16)f1.w;
    *(bf16x8*)(cbh + (size_t)k * D + i) = h;

    float ss = f0.x*f0.x + f0.y*f0.y + f0.z*f0.z + f0.w*f0.w
             + f1.x*f1.x + f1.y*f1.y + f1.z*f1.z + f1.w*f1.w;
    #pragma unroll
    for (int m = 16; m >= 1; m >>= 1) ss += __shfl_xor(ss, m, 32);

    __shared__ float red[2];
    if ((tid & 31) == 0) red[tid >> 5] = ss;
    __syncthreads();
    if (tid == 0) {
        cnorm[k] = red[0] + red[1];
        if (k == 0) *lossAcc = 0.0f;        // re-zero every call (graph replay)
    }
}

// ---------------------------------------------------------------------------
// Helpers
// ---------------------------------------------------------------------------

// order-preserving float->uint key with code index packed in the low 12 bits
__device__ __forceinline__ unsigned sortKey(float s, unsigned idx) {
    unsigned u = __float_as_uint(s);
    u ^= ((unsigned)((int)u >> 31)) | 0x80000000u;
    return (u & 0xFFFFF000u) | idx;
}

// build one resident A-fragment set: 16 rows x full D (16 x v16bf = 128 VGPRs)
__device__ __forceinline__ void loadA(const float* __restrict__ x, int row, int hf,
                                      bf16x16* aF) {
    #pragma unroll
    for (int t = 0; t < 16; ++t) {
        const int c0 = t * 32 + hf * 8;
        const float4* p0 = (const float4*)(x + (size_t)row * D + c0);
        const float4* p1 = (const float4*)(x + (size_t)row * D + c0 + 16);
        float4 f0 = p0[0], f1 = p0[1], f2 = p1[0], f3 = p1[1];
        bf16x16 a;
        a[0]  = (__bf16)f0.x; a[1]  = (__bf16)f0.y; a[2]  = (__bf16)f0.z; a[3]  = (__bf16)f0.w;
        a[4]  = (__bf16)f1.x; a[5]  = (__bf16)f1.y; a[6]  = (__bf16)f1.z; a[7]  = (__bf16)f1.w;
        a[8]  = (__bf16)f2.x; a[9]  = (__bf16)f2.y; a[10] = (__bf16)f2.z; a[11] = (__bf16)f2.w;
        a[12] = (__bf16)f3.x; a[13] = (__bf16)f3.y; a[14] = (__bf16)f3.z; a[15] = (__bf16)f3.w;
        aF[t] = a;
    }
}

__device__ __forceinline__ bf16x16 loadB(const __bf16* Bp, int t, int r16, int hf) {
    const bf16x8 b0 = *(const bf16x8*)(Bp + r16 * LDB + t * 32 + hf * 8);
    const bf16x8 b1 = *(const bf16x8*)(Bp + r16 * LDB + t * 32 + 16 + hf * 8);
    return __builtin_shufflevector(b0, b1,
        0, 1, 2, 3, 4, 5, 6, 7, 8, 9, 10, 11, 12, 13, 14, 15);
}

// async-DMA stage one 16-code chunk (16 x 512 bf16) global -> LDS, no VGPR data
__device__ __forceinline__ void stageAsync(const __bf16* __restrict__ cbh,
                                           __bf16* dst, int kk, int tid) {
    #pragma unroll
    for (int i = 0; i < 4; ++i) {
        const int cid = tid + 256 * i;        // 1024 b128 pieces
        const int br  = cid >> 6;
        const int bc  = (cid & 63) << 3;
        // low 32 bits of a generic shared pointer == LDS byte address
        unsigned lds  = (unsigned)(size_t)(dst + br * LDB + bc);
        unsigned voff = (unsigned)(((kk * 16 + br) * D + bc) * sizeof(__bf16));
        asm volatile("global_load_async_to_lds_b128 %0, %1, %2 offset:0"
                     :: "v"(lds), "v"(voff), "s"(cbh) : "memory");
    }
}

__device__ __forceinline__ void waitAsync0() {
    asm volatile("s_wait_asynccnt 0x0" ::: "memory");
}

// ---------------------------------------------------------------------------
// Kernel 2: main VQ — WMMA bf16 scores, async double-buffered B, argmin,
//           gather + loss epilogue. 8 wave32 waves, 32 rows each (2 tiles).
// ---------------------------------------------------------------------------
__global__ void __launch_bounds__(256) vq_main(const float* __restrict__ x,
                                               const float* __restrict__ cb,
                                               const __bf16* __restrict__ cbh,
                                               const float* __restrict__ cnorm,
                                               float* __restrict__ out,
                                               float* __restrict__ lossAcc) {
    __shared__ __align__(16) __bf16 Bs[2][16 * LDB];  // double-buffered B tiles
    __shared__ float cnormS[K];
    __shared__ int   idxS[BM];
    __shared__ float redS[8];

    const int tid  = threadIdx.x;
    const int w    = tid >> 5;        // wave 0..7
    const int lane = tid & 31;
    const int hf   = lane >> 4;       // fragment lane-half
    const int r16  = lane & 15;
    const int blockRow = blockIdx.x * BM;

    for (int i = tid; i < K; i += 256) cnormS[i] = cnorm[i];

    // two resident A tiles per wave: rows [w*32, w*32+16) and [w*32+16, w*32+32)
    const int rowA0 = blockRow + w * 32 + r16;
    bf16x16 aF0[16], aF1[16];
    loadA(x, rowA0,      hf, aF0);
    loadA(x, rowA0 + 16, hf, aF1);

    unsigned best0[8], best1[8];
    #pragma unroll
    for (int j = 0; j < 8; ++j) { best0[j] = 0xFFFFFFFFu; best1[j] = 0xFFFFFFFFu; }

    stageAsync(cbh, &Bs[0][0], 0, tid);           // prologue: chunk 0 in flight

    for (int kk = 0; kk < NCHUNK; ++kk) {
        waitAsync0();                              // my chunk-kk pieces landed
        __syncthreads();                           // everyone's pieces landed
        if (kk + 1 < NCHUNK)                       // prefetch next chunk
            stageAsync(cbh, &Bs[(kk + 1) & 1][0], kk + 1, tid);

        const __bf16* Bp = &Bs[kk & 1][0];
        f32x8 c0 = {}, c1 = {};
        bf16x16 bcur = loadB(Bp, 0, r16, hf);
        #pragma unroll
        for (int t = 0; t < 16; ++t) {             // software-pipelined B frags
            bf16x16 bnext;
            if (t < 15) bnext = loadB(Bp, t + 1, r16, hf);
            c0 = __builtin_amdgcn_wmma_f32_16x16x32_bf16(
                     false, aF0[t], false, bcur, (short)0, c0, false, false);
            c1 = __builtin_amdgcn_wmma_f32_16x16x32_bf16(
                     false, aF1[t], false, bcur, (short)0, c1, false, false);
            bcur = bnext;
        }

        // dist = ||c||^2 - 2*dot ; min+argmin via packed-key u32 butterflies
        const int      kbase = kk * 16;
        const float    cn    = cnormS[kbase + r16];
        const unsigned myk   = (unsigned)(kbase + r16);
        #pragma unroll
        for (int j = 0; j < 8; ++j) {
            unsigned k0 = sortKey(cn - 2.0f * c0[j], myk);
            unsigned k1 = sortKey(cn - 2.0f * c1[j], myk);
            #pragma unroll
            for (int m = 1; m < 16; m <<= 1) {
                k0 = min(k0, (unsigned)__shfl_xor((int)k0, m, 16));
                k1 = min(k1, (unsigned)__shfl_xor((int)k1, m, 16));
            }
            best0[j] = min(best0[j], k0);
            best1[j] = min(best1[j], k1);
        }
    }

    if (r16 == 0) {
        #pragma unroll
        for (int j = 0; j < 8; ++j) {
            idxS[w * 32 +      hf * 8 + j] = (int)(best0[j] & 0xFFFu);
            idxS[w * 32 + 16 + hf * 8 + j] = (int)(best1[j] & 0xFFFu);
        }
    }
    __syncthreads();

    // Epilogue: gather exact fp32 code rows -> quantized output, fp32 loss
    float accL = 0.0f;
    const int c0i = tid * 2;
    for (int r = 0; r < BM; ++r) {
        const int    kq = idxS[r];
        const size_t gr = (size_t)(blockRow + r);
        float2 q  = *(const float2*)(cb + (size_t)kq * D + c0i);
        float2 xx = *(const float2*)(x  + gr * D + c0i);
        *(float2*)(out + gr * D + c0i) = q;
        const float dx = q.x - xx.x, dy = q.y - xx.y;
        accL += dx * dx + dy * dy;
    }
    out[(size_t)NQ + 1 + blockRow + tid] = (float)idxS[tid];  // BM == blockDim

    #pragma unroll
    for (int m = 16; m >= 1; m >>= 1) accL += __shfl_xor(accL, m, 32);
    if (lane == 0) redS[w] = accL;
    __syncthreads();
    if (tid == 0) {
        float t = 0.0f;
        #pragma unroll
        for (int i = 0; i < 8; ++i) t += redS[i];
        atomicAdd(lossAcc, t);
    }
}

// ---------------------------------------------------------------------------
// Kernel 3: loss = (COMMITMENT_COST + 1) * mean((q - x)^2)
// ---------------------------------------------------------------------------
__global__ void vq_fin(const float* __restrict__ lossAcc, float* __restrict__ out) {
    out[NQ] = 1.25f * (*lossAcc) / (float)NQ;
}

extern "C" void kernel_launch(void* const* d_in, const int* in_sizes, int n_in,
                              void* d_out, int out_size, void* d_ws, size_t ws_size,
                              hipStream_t stream) {
    const float* x  = (const float*)d_in[0];   // inputs  [64,512,512] fp32
    const float* cb = (const float*)d_in[1];   // codebook [4096,512] fp32
    float* out = (float*)d_out;

    // workspace layout: cnorm (16KB) | codebook bf16 (4MB) | loss acc (4B)
    float*  cnorm   = (float*)d_ws;
    __bf16* cbh     = (__bf16*)((char*)d_ws + (size_t)K * sizeof(float));
    float*  lossAcc = (float*)((char*)d_ws + (size_t)K * sizeof(float)
                                           + (size_t)K * D * sizeof(__bf16));

    vq_prep<<<K, 64, 0, stream>>>(cb, cnorm, cbh, lossAcc);
    vq_main<<<NROWS / BM, 256, 0, stream>>>(x, cb, cbh, cnorm, out, lossAcc);
    vq_fin<<<1, 1, 0, stream>>>(lossAcc, out);
}
